// RetrievalHead_7782480740812
// MI455X (gfx1250) — matrix operand order repaired and checked by reading, not compiled, not run on hardware
//
#include <hip/hip_runtime.h>

typedef __attribute__((ext_vector_type(16))) _Float16 v16h;
typedef __attribute__((ext_vector_type(8)))  float    v8f;
typedef __attribute__((ext_vector_type(4)))  int      v4i;
typedef __attribute__((ext_vector_type(8)))  int      v8i;

#define NQ   64
#define NC   2048
#define NVOL 2112
#define NCODE 256
#define EPSF 1e-6f

// ---------------------------------------------------------------------------
// Repack conv weights f32 [Cout][Cin][3][3][3] -> f16 [Cout][Kpad] row major.
// ---------------------------------------------------------------------------
__global__ void pack_w_kernel(const float* __restrict__ W, _Float16* __restrict__ out,
                              int Cout, int K, int Kpad) {
    int i = blockIdx.x * blockDim.x + threadIdx.x;
    if (i >= Cout * Kpad) return;
    int m = i / Kpad, k = i % Kpad;
    out[i] = (_Float16)((k < K) ? W[(size_t)m * K + k] : 0.f);
}

// ---------------------------------------------------------------------------
// Layer 1: f32 input (1ch, 32^3) -> f16 (32ch, 16^3), stride 2, SAME (hi pad).
// K=27 padded to 32 -> single 16x16x32 WMMA per tile. One wave per tile.
// ---------------------------------------------------------------------------
__global__ __launch_bounds__(32)
void conv1_wmma_kernel(const float* __restrict__ q, const float* __restrict__ c,
                       const _Float16* __restrict__ Wh, const float* __restrict__ bias,
                       _Float16* __restrict__ out) {
    const int I = 32, SP = 4096;
    int wg  = blockIdx.x;
    int mt  = wg & 1;
    int nt  = (wg >> 1) & 255;
    int vol = wg >> 9;
    int lane = threadIdx.x & 31;
    int lrow = lane & 15, hi = lane >> 4;

    const float* in = (vol < NQ) ? (q + (size_t)vol * 32768)
                                 : (c + (size_t)(vol - NQ) * 32768);
    int pos = nt * 16 + lrow;
    int od = pos >> 8, oh = (pos >> 4) & 15, ow = pos & 15;

    v16h bfrag;
#pragma unroll
    for (int p = 0; p < 8; ++p) {
#pragma unroll
        for (int h = 0; h < 2; ++h) {
            int k = (p < 4 ? 2 * p : 8 + 2 * p) + 8 * hi + h;
            float v = 0.f;
            if (k < 27) {
                int kd = k / 9, kh = (k % 9) / 3, kw = k % 3;
                int id = od * 2 + kd, ih = oh * 2 + kh, iw = ow * 2 + kw;
                if (id < I && ih < I && iw < I)
                    v = in[(id * I + ih) * I + iw];
            }
            bfrag[2 * p + h] = (_Float16)v;
        }
    }
    v16h afrag;
    int m = mt * 16 + lrow;
#pragma unroll
    for (int p = 0; p < 8; ++p) {
        int kb = (p < 4 ? 2 * p : 8 + 2 * p) + 8 * hi;
        afrag[2 * p]     = Wh[m * 32 + kb];
        afrag[2 * p + 1] = Wh[m * 32 + kb + 1];
    }
    v8f acc = {};
    acc = __builtin_amdgcn_wmma_f32_16x16x32_f16(false, afrag, false, bfrag,
                                                 (short)0, acc, false, false);
#pragma unroll
    for (int r = 0; r < 8; ++r) {
        int mm = mt * 16 + r + 8 * hi;
        float v = fmaxf(acc[r] + bias[mm], 0.f);
        out[((size_t)vol * 32 + mm) * SP + pos] = (_Float16)v;
    }
}

// ---------------------------------------------------------------------------
// Mid layers (2,3) as LDS-staged implicit GEMM. One block per (volume, od
// slice). The 3-deep input slab [CIN][3][II][II] is staged into LDS with
// async global->LDS copies; each of the COUT/16 waves owns one M tile, reuses
// one A fragment across all N tiles of the slice (4 WMMAs per A-load for L2).
// ---------------------------------------------------------------------------
template <int CIN, int II, int OO, int COUT>
__global__ __launch_bounds__((COUT / 16) * 32)
void conv_mid_kernel(const _Float16* __restrict__ in, const _Float16* __restrict__ Wh,
                     const float* __restrict__ bias, _Float16* __restrict__ out) {
    constexpr int K    = CIN * 27;
    constexpr int MT   = COUT / 16;
    constexpr int NT   = (OO * OO) / 16;       // N tiles per od slice
    constexpr int SLAB = CIN * 3 * II * II;    // halves
    constexpr int SP   = OO * OO * OO;
    __shared__ _Float16 slab[SLAB];

    int od  = blockIdx.x % OO;
    int vol = blockIdx.x / OO;
    int t    = threadIdx.x;
    int lane = t & 31;
    int wave = t >> 5;                          // == M tile
    int lrow = lane & 15, hi = lane >> 4;

    const _Float16* inv = in + (size_t)vol * CIN * II * II * II;

    // ---- async stage: slab[cin][dz][ih][iw] <- in[vol][cin][2*od+dz][:][:] ----
    constexpr int CHUNKS = SLAB / 8;            // 16B chunks
    for (int ch = t; ch < CHUNKS; ch += MT * 32) {
        int plane = ch / (II * II / 8);         // cin*3 + dz
        int off   = ch % (II * II / 8);
        int cin = plane / 3, dz = plane % 3;
        int id  = od * 2 + dz;
        if (id < II) {                          // hi-side SAME pad row never read
            unsigned ldsoff = (unsigned)(uintptr_t)&slab[ch * 8];
            int goff = ((cin * II + id) * II * II) * 2 + off * 16;
            asm volatile("global_load_async_to_lds_b128 %0, %1, %2"
                         :: "v"(ldsoff), "v"(goff), "s"(inv) : "memory");
        }
    }
    asm volatile("s_wait_asynccnt 0" ::: "memory");
    __syncthreads();

    const _Float16* wrow = Wh + (size_t)(wave * 16 + lrow) * K;

    v8f acc[NT] = {};
    for (int k0 = 0; k0 < K; k0 += 32) {
        v16h afrag;
#pragma unroll
        for (int p = 0; p < 8; ++p) {
            int ka = k0 + (p < 4 ? 2 * p : 8 + 2 * p) + 8 * hi;
            afrag[2 * p]     = wrow[ka];
            afrag[2 * p + 1] = wrow[ka + 1];
        }
#pragma unroll
        for (int nt2 = 0; nt2 < NT; ++nt2) {
            int local = nt2 * 16 + lrow;
            int oh = local / OO, ow = local % OO;
            v16h bfrag;
#pragma unroll
            for (int p = 0; p < 8; ++p) {
#pragma unroll
                for (int h = 0; h < 2; ++h) {
                    int k   = k0 + (p < 4 ? 2 * p : 8 + 2 * p) + 8 * hi + h;
                    int cin = k / 27;
                    int rem = k % 27;
                    int kd = rem / 9, kh = (rem % 9) / 3, kw = rem % 3;
                    int id = od * 2 + kd, ih = oh * 2 + kh, iw = ow * 2 + kw;
                    _Float16 v = (_Float16)0.f;
                    if (id < II && ih < II && iw < II)
                        v = slab[((cin * 3 + kd) * II + ih) * II + iw];
                    bfrag[2 * p + h] = v;
                }
            }
            acc[nt2] = __builtin_amdgcn_wmma_f32_16x16x32_f16(
                false, afrag, false, bfrag, (short)0, acc[nt2], false, false);
        }
    }
#pragma unroll
    for (int nt2 = 0; nt2 < NT; ++nt2) {
#pragma unroll
        for (int r = 0; r < 8; ++r) {
            int mm  = wave * 16 + r + 8 * hi;
            int pos = od * OO * OO + nt2 * 16 + lrow;
            float v = fmaxf(acc[nt2][r] + bias[mm], 0.f);
            out[((size_t)vol * COUT + mm) * SP + pos] = (_Float16)v;
        }
    }
}

// ---------------------------------------------------------------------------
// Layer 4 + global mean pool. Two volumes per block (packed into one N tile),
// both staged fully into LDS (32KB) via async copies; 16 waves = 16 M tiles.
// relu + mean over the 8 positions via __shfl_xor from the accumulator.
// ---------------------------------------------------------------------------
__global__ __launch_bounds__(512)
void conv4_pool_wmma_kernel(const _Float16* __restrict__ in, const _Float16* __restrict__ Wh,
                            const float* __restrict__ bias, float* __restrict__ codes) {
    const int CIN = 128, II = 4, K = 3456;
    __shared__ _Float16 slab[2 * CIN * II * II * II];   // 16384 halves = 32KB

    int vp = blockIdx.x;
    int t = threadIdx.x;
    int lane = t & 31;
    int mt = t >> 5;
    int lrow = lane & 15, hi = lane >> 4;

    const _Float16* inv = in + (size_t)vp * 2 * CIN * II * II * II;

    // flat 32KB copy: both volumes are contiguous in act3
    for (int ch = t; ch < 2048; ch += 512) {
        unsigned ldsoff = (unsigned)(uintptr_t)&slab[ch * 8];
        asm volatile("global_load_async_to_lds_b128 %0, %1, %2"
                     :: "v"(ldsoff), "v"(ch * 16), "s"(inv) : "memory");
    }
    asm volatile("s_wait_asynccnt 0" ::: "memory");
    __syncthreads();

    int vsel = lrow >> 3;
    int pos  = lrow & 7;
    int od = pos >> 2, oh = (pos >> 1) & 1, ow = pos & 1;
    const _Float16* wrow = Wh + (size_t)(mt * 16 + lrow) * K;
    const _Float16* svol = &slab[vsel * CIN * II * II * II];

    v8f acc = {};
    for (int k0 = 0; k0 < K; k0 += 32) {
        v16h afrag, bfrag;
#pragma unroll
        for (int p = 0; p < 8; ++p) {
            int ka = k0 + (p < 4 ? 2 * p : 8 + 2 * p) + 8 * hi;
            afrag[2 * p]     = wrow[ka];
            afrag[2 * p + 1] = wrow[ka + 1];
#pragma unroll
            for (int h = 0; h < 2; ++h) {
                int k   = k0 + (p < 4 ? 2 * p : 8 + 2 * p) + 8 * hi + h;
                int cin = k / 27;
                int rem = k % 27;
                int kd = rem / 9, kh = (rem % 9) / 3, kw = rem % 3;
                int id = od * 2 + kd, ih = oh * 2 + kh, iw = ow * 2 + kw;
                _Float16 v = (_Float16)0.f;
                if (id < II && ih < II && iw < II)
                    v = svol[((cin * II + id) * II + ih) * II + iw];
                bfrag[2 * p + h] = v;
            }
        }
        acc = __builtin_amdgcn_wmma_f32_16x16x32_f16(false, afrag, false, bfrag,
                                                     (short)0, acc, false, false);
    }
#pragma unroll
    for (int r = 0; r < 8; ++r) {
        int mm = mt * 16 + r + 8 * hi;
        float v = fmaxf(acc[r] + bias[mm], 0.f);
#pragma unroll
        for (int o = 1; o < 8; o <<= 1)
            v += __shfl_xor(v, o, 32);
        if ((lane & 7) == 0)
            codes[(size_t)(vp * 2 + vsel) * NCODE + mm] = v * 0.125f;
    }
}

// ---------------------------------------------------------------------------
__global__ void copy_q_kernel(const float* __restrict__ codes, float* __restrict__ outq) {
    int i = blockIdx.x * blockDim.x + threadIdx.x;
    if (i < NQ * NCODE) outq[i] = codes[i];
}

// ---------------------------------------------------------------------------
// Pairwise distances + argmin. One block (256 thr) per query. The query row
// is pulled into LDS by the Tensor Data Mover: hand-built D# (group0 128b +
// group1 256b per CDNA5 ISA) issued from wave 0, drained
// with s_wait_tensorcnt.
// ---------------------------------------------------------------------------
__global__ __launch_bounds__(256)
void dist_argmin_kernel(const float* __restrict__ codes, float* __restrict__ dout) {
    __shared__ float qs[NCODE];
    __shared__ float smin[256];
    __shared__ int   sidx[256];

    int qi = blockIdx.x;
    int t  = threadIdx.x;
    const float* qrow = codes + (size_t)qi * NCODE;

    if (t < 32) {   // one TDM issue per block (EXEC ignored; wave-level op)
        unsigned long long ga = (unsigned long long)(uintptr_t)qrow;
        unsigned ldsa = (unsigned)(uintptr_t)&qs[0];
        v4i g0;
        g0[0] = 1;                                            // count=1, user D#
        g0[1] = (int)ldsa;                                    // lds_addr
        g0[2] = (int)(ga & 0xffffffffu);                      // global_addr[31:0]
        g0[3] = (int)(((ga >> 32) & 0x01ffffffu) | 0x80000000u); // ga[56:32] | type=2
        v8i g1;
        g1[0] = 0x00020000;   // data_size=4B, no multicast/iterate/pad
        g1[1] = 0x01000000;   // tensor_dim0 = 256 (low 16 in [31:16])
        g1[2] = 0x00010000;   // tensor_dim1 = 1
        g1[3] = 0x01000000;   // tile_dim0 = 256
        g1[4] = 0x00000001;   // tile_dim1 = 1
        g1[5] = 256;          // tensor_dim0_stride = 256
        g1[6] = 0;
        g1[7] = 0;
        asm volatile("tensor_load_to_lds %0, %1" :: "s"(g0), "s"(g1) : "memory");
    }
    __builtin_amdgcn_s_wait_tensorcnt(0);
    __syncthreads();

    float best = 3.4e38f;
    int   bidx = 0;
    for (int ci = t; ci < NC; ci += 256) {
        const float* crow = codes + (size_t)(NQ + ci) * NCODE;
        __builtin_prefetch(crow + 64 * NCODE, 0, 0);          // -> global_prefetch_b8
        float s = 0.f;
#pragma unroll 8
        for (int k = 0; k < NCODE; ++k) {
            float d = qs[k] - crow[k] + EPSF;
            s = fmaf(d, d, s);
        }
        float dist = __builtin_sqrtf(s);
        dout[NQ * NCODE + (size_t)qi * NC + ci] = dist;
        if (dist < best) { best = dist; bidx = ci; }
    }
    smin[t] = best; sidx[t] = bidx;
    __syncthreads();
    for (int o = 128; o > 0; o >>= 1) {
        if (t < o && smin[t + o] < smin[t]) { smin[t] = smin[t + o]; sidx[t] = sidx[t + o]; }
        __syncthreads();
    }
    if (t == 0)
        ((int*)dout)[NQ * NCODE + NQ * NC + qi] = sidx[0];
}

// ---------------------------------------------------------------------------
extern "C" void kernel_launch(void* const* d_in, const int* in_sizes, int n_in,
                              void* d_out, int out_size, void* d_ws, size_t ws_size,
                              hipStream_t stream) {
    (void)in_sizes; (void)n_in; (void)out_size; (void)ws_size;
    const float* q  = (const float*)d_in[0];
    const float* c  = (const float*)d_in[1];
    const float* W1 = (const float*)d_in[2]; const float* b1 = (const float*)d_in[3];
    const float* W2 = (const float*)d_in[4]; const float* b2 = (const float*)d_in[5];
    const float* W3 = (const float*)d_in[6]; const float* b3 = (const float*)d_in[7];
    const float* W4 = (const float*)d_in[8]; const float* b4 = (const float*)d_in[9];
    float* out = (float*)d_out;

    char* ws = (char*)d_ws;
    _Float16* W1h  = (_Float16*)ws;  ws += (size_t)32  * 32   * 2;
    _Float16* W2h  = (_Float16*)ws;  ws += (size_t)64  * 864  * 2;
    _Float16* W3h  = (_Float16*)ws;  ws += (size_t)128 * 1728 * 2;
    _Float16* W4h  = (_Float16*)ws;  ws += (size_t)256 * 3456 * 2;
    _Float16* act1 = (_Float16*)ws;  ws += (size_t)NVOL * 32  * 4096 * 2;
    _Float16* act2 = (_Float16*)ws;  ws += (size_t)NVOL * 64  * 512  * 2;
    _Float16* act3 = (_Float16*)ws;  ws += (size_t)NVOL * 128 * 64   * 2;
    float*    codes = (float*)ws;    // NVOL * 256 f32

    pack_w_kernel<<<(32 * 32 + 255) / 256, 256, 0, stream>>>(W1, W1h, 32, 27, 32);
    pack_w_kernel<<<(64 * 864 + 255) / 256, 256, 0, stream>>>(W2, W2h, 64, 864, 864);
    pack_w_kernel<<<(128 * 1728 + 255) / 256, 256, 0, stream>>>(W3, W3h, 128, 1728, 1728);
    pack_w_kernel<<<(256 * 3456 + 255) / 256, 256, 0, stream>>>(W4, W4h, 256, 3456, 3456);

    conv1_wmma_kernel<<<NVOL * 512, 32, 0, stream>>>(q, c, W1h, b1, act1);
    conv_mid_kernel<32, 16, 8, 64><<<NVOL * 8, 128, 0, stream>>>(act1, W2h, b2, act2);
    conv_mid_kernel<64, 8, 4, 128><<<NVOL * 4, 256, 0, stream>>>(act2, W3h, b3, act3);
    conv4_pool_wmma_kernel<<<NVOL / 2, 512, 0, stream>>>(act3, W4h, b4, codes);

    copy_q_kernel<<<(NQ * NCODE + 255) / 256, 256, 0, stream>>>(codes, out);
    dist_argmin_kernel<<<NQ, 256, 0, stream>>>(codes, out);
}
